// StructureModule_15539191677790
// MI455X (gfx1250) — matrix operand order repaired and checked by reading, not compiled, not run on hardware
//
#include <hip/hip_runtime.h>
#include <hip/hip_bf16.h>

// ============================================================================
// StructureModule for MI455X (gfx1250, wave32, WMMA).
//
// Roofline: z (134MB) dominates traffic; all matmuls run as bf16 WMMA
// (v_wmma_f32_16x16x32_bf16) with f32 accumulation. Row-LayerNorm is fused
// into the GEMM A-tile staging (K == full row length for every LN'd GEMM).
// The LN(z)@lin_b_w bias GEMM is recomputed per attention block (8 x 134MB
// = ~46us at 23.3 TB/s) to keep workspace at ~23MB instead of 100MB+.
//
// GEMM contract (all call sites obey): K % 32 == 0, M % 16 == 0,
// A rows and B (for BT path) 16-byte aligned with ld % 4 == 0.
//
// Input flattening assumption (setup_inputs() dict order, params in
// insertion order): 0:s 1:z 2:aatype 3:mask 4..131:params leaves
// 132:default_frames 133:group_idx 134:atom_mask 135:lit_positions.
// ============================================================================

typedef __attribute__((ext_vector_type(16))) __bf16 v16bf_t;
typedef __attribute__((ext_vector_type(8)))  float  v8f_t;

#define NRES 512
#define CS   384
#define CZ   128
#define NHEADS 12
#define HDIM 32
#define NBLK 8
#define CRES 128
#define MZ   (NRES * NRES)   // 262144 rows of the z-GEMM

__device__ __forceinline__ unsigned bfbits(float f) {
  unsigned u = __builtin_bit_cast(unsigned, f);
  u += 0x7FFFu + ((u >> 16) & 1u);             // round-to-nearest-even
  return (u >> 16) & 0xFFFFu;
}
__device__ __forceinline__ unsigned pack2(float lo, float hi) {
  return (bfbits(hi) << 16) | bfbits(lo);
}

// A_OP: 0=identity 1=relu 2=layernorm(g,b).  OUT_OP: 0=id 1=relu 2=sigmoid.
// BT:   B is stored N x K (use transposed).   OUTT: store C transposed.
// C[m][n] = out_op( alpha * (a_op(A)[m,:] @ B[:,n]) + biasv[n] + Cadd[m][n] )
template <int A_OP, int OUT_OP, bool BT, bool OUTT>
__global__ __launch_bounds__(256) void gemm_wmma(
    const float* __restrict__ A, long sA,
    const float* __restrict__ Bm, long sB,
    const float* __restrict__ biasv,
    const float* __restrict__ Cadd, long sAdd, int ldadd,
    const float* __restrict__ lng, const float* __restrict__ lnb,
    float* __restrict__ C, long sC,
    int M, int K, int Ncols, int lda, int ldb, int ldc, float alpha)
{
  __shared__ __bf16 As[16 * 512];        // A tile, bf16, post-op, [r][k] (16KB)
  __shared__ __bf16 Bs[128 * 32];        // B k-slab, [n][kk]             (8KB)
  __shared__ float  redA[256], redB[256];
  __shared__ float  mu[16], rs[16];

  const int t    = threadIdx.x;
  const int lane = t & 31;
  const int wave = t >> 5;
  const int mt   = blockIdx.x;           // 16-row M tile
  const int nb   = blockIdx.y * 128;     // 128-col N group (8 waves x 16)
  const int bat  = blockIdx.z;

  const float* Ab = A  + (long)bat * sA;
  const float* Bb = Bm + (long)bat * sB;
  float*       Cb = C  + (long)bat * sC;

  // A staging: each thread owns a contiguous chunk of one row.
  const int  r     = t >> 4;             // staging row 0..15
  const int  l16   = t & 15;
  const int  chunk = K >> 4;             // K/16 floats per thread (even)
  const int  k0    = l16 * chunk;
  const long arow  = (long)(mt * 16 + r) * lda;   // M % 16 == 0 guaranteed
  const bool vec4  = (chunk & 3) == 0;

  if (A_OP == 2) {                       // LN stats over the full row
    float s1 = 0.f, s2 = 0.f;
    if (vec4) {
      for (int j = 0; j < chunk; j += 4) {
        float4 v4 = *(const float4*)(Ab + arow + k0 + j);
        s1 += v4.x + v4.y + v4.z + v4.w;
        s2 += v4.x * v4.x + v4.y * v4.y + v4.z * v4.z + v4.w * v4.w;
      }
    } else {
      for (int j = 0; j < chunk; ++j) { float v = Ab[arow + k0 + j]; s1 += v; s2 += v * v; }
    }
    redA[t] = s1; redB[t] = s2;
    __syncthreads();
    if (t < 16) {
      float a = 0.f, b = 0.f;
      for (int j = 0; j < 16; ++j) { a += redA[t * 16 + j]; b += redB[t * 16 + j]; }
      float m = a / (float)K;
      float var = b / (float)K - m * m;
      mu[t] = m; rs[t] = rsqrtf(var + 1e-5f);
    }
    __syncthreads();
  }

  // Stage A as bf16 (packed pair stores).
  {
    unsigned* As32 = (unsigned*)As;
    const int base = (r * K + k0) >> 1;  // dword index (k0, K even)
    if (vec4) {
      for (int j = 0; j < chunk; j += 4) {
        float4 v4 = *(const float4*)(Ab + arow + k0 + j);
        float v0 = v4.x, v1 = v4.y, v2 = v4.z, v3 = v4.w;
        if (A_OP == 1) {
          v0 = fmaxf(v0, 0.f); v1 = fmaxf(v1, 0.f);
          v2 = fmaxf(v2, 0.f); v3 = fmaxf(v3, 0.f);
        }
        if (A_OP == 2) {
          int k = k0 + j;
          v0 = (v0 - mu[r]) * rs[r] * lng[k + 0] + lnb[k + 0];
          v1 = (v1 - mu[r]) * rs[r] * lng[k + 1] + lnb[k + 1];
          v2 = (v2 - mu[r]) * rs[r] * lng[k + 2] + lnb[k + 2];
          v3 = (v3 - mu[r]) * rs[r] * lng[k + 3] + lnb[k + 3];
        }
        As32[base + (j >> 1)]     = pack2(v0, v1);
        As32[base + (j >> 1) + 1] = pack2(v2, v3);
      }
    } else {
      for (int j = 0; j < chunk; j += 2) {
        float v0 = Ab[arow + k0 + j], v1 = Ab[arow + k0 + j + 1];
        if (A_OP == 1) { v0 = fmaxf(v0, 0.f); v1 = fmaxf(v1, 0.f); }
        if (A_OP == 2) {
          int k = k0 + j;
          v0 = (v0 - mu[r]) * rs[r] * lng[k + 0] + lnb[k + 0];
          v1 = (v1 - mu[r]) * rs[r] * lng[k + 1] + lnb[k + 1];
        }
        As32[base + (j >> 1)] = pack2(v0, v1);
      }
    }
  }
  __syncthreads();

  v8f_t acc;
#pragma unroll
  for (int i = 0; i < 8; ++i) acc[i] = 0.f;

  const int half = lane >> 4;
  const int ml   = lane & 15;
  const int myn  = nb + wave * 16;       // this wave's N tile base

  // Prefetch the additive-C tile (attention bias) while we do the math.
  if (Cadd) {
#pragma unroll
    for (int rr = 0; rr < 8; ++rr)
      __builtin_prefetch(
          &Cadd[(long)bat * sAdd + (long)(mt * 16 + rr + 8 * half) * ldadd + myn + ml],
          0, 3);
  }

  // B staging assignment: thread owns 16 consecutive kk of one n column.
  const int bn   = t >> 1;               // 0..127
  const int bk0  = (t & 1) * 16;         // 0 or 16
  const int ngB  = nb + bn;
  unsigned* Bs32 = (unsigned*)Bs;
  const int bbase = (bn * 32 + bk0) >> 1;

  for (int kb = 0; kb < K; kb += 32) {
    __syncthreads();                     // previous Bs fully consumed
    if (ngB < Ncols) {
      if (BT) {                          // B is N x K: contiguous along kk
        const float* src = Bb + (long)ngB * ldb + kb + bk0;
        float4 a0 = *(const float4*)(src + 0);
        float4 a1 = *(const float4*)(src + 4);
        float4 a2 = *(const float4*)(src + 8);
        float4 a3 = *(const float4*)(src + 12);
        Bs32[bbase + 0] = pack2(a0.x, a0.y); Bs32[bbase + 1] = pack2(a0.z, a0.w);
        Bs32[bbase + 2] = pack2(a1.x, a1.y); Bs32[bbase + 3] = pack2(a1.z, a1.w);
        Bs32[bbase + 4] = pack2(a2.x, a2.y); Bs32[bbase + 5] = pack2(a2.z, a2.w);
        Bs32[bbase + 6] = pack2(a3.x, a3.y); Bs32[bbase + 7] = pack2(a3.z, a3.w);
      } else {                           // B is K x N: stride-ldb gathers
#pragma unroll
        for (int j = 0; j < 16; j += 2) {
          float v0 = Bb[(long)(kb + bk0 + j)     * ldb + ngB];
          float v1 = Bb[(long)(kb + bk0 + j + 1) * ldb + ngB];
          Bs32[bbase + (j >> 1)] = pack2(v0, v1);
        }
      }
    } else {
#pragma unroll
      for (int j = 0; j < 8; ++j) Bs32[bbase + j] = 0u;
    }
    __syncthreads();

    // Fragments use k = 16*half + e for both A and B; both are contiguous
    // LDS runs so the compiler can use ds_load_b128.
    v16bf_t af, bfv;
    const __bf16* ap = As + ml * K + kb;
    const __bf16* bp = Bs + (wave * 16 + ml) * 32 + 16 * half;
#pragma unroll
    for (int e = 0; e < 16; ++e) {
      int kl = e + 16 * half;                         // A: K map
      af[e]  = ap[kl];
      bfv[e] = bp[e];                                 // B: K = e + 16*half
    }
    acc = __builtin_amdgcn_wmma_f32_16x16x32_bf16(false, af, false, bfv,
                                                  (short)0, acc, false, false);
  }

  // C/D layout: VGPR rr -> M = rr + 8*half, lane%16 = N.
  const int ng = myn + ml;
  if (ng < Ncols) {
#pragma unroll
    for (int rr = 0; rr < 8; ++rr) {
      int mg = mt * 16 + rr + 8 * half;
      float v = alpha * acc[rr];
      if (biasv) v += biasv[ng];
      if (Cadd)  v += Cadd[(long)bat * sAdd + (long)mg * ldadd + ng];
      if (OUT_OP == 1) v = fmaxf(v, 0.f);
      if (OUT_OP == 2) v = 1.f / (1.f + __expf(-v));
      if (OUTT) Cb[(long)ng * ldc + mg] = v;
      else      Cb[(long)mg * ldc + ng] = v;
    }
  }
}

// ---------------------------------------------------------------------------
__global__ __launch_bounds__(128) void ln_rows(const float* __restrict__ x,
                                               const float* __restrict__ g,
                                               const float* __restrict__ b,
                                               float* __restrict__ y, int C)
{
  __shared__ float s1[128], s2[128];
  const int row = blockIdx.x;
  const float* xr = x + (long)row * C;
  float a = 0.f, q = 0.f;
  for (int k = threadIdx.x; k < C; k += 128) { float v = xr[k]; a += v; q += v * v; }
  s1[threadIdx.x] = a; s2[threadIdx.x] = q;
  __syncthreads();
  for (int o = 64; o; o >>= 1) {
    if (threadIdx.x < o) { s1[threadIdx.x] += s1[threadIdx.x + o]; s2[threadIdx.x] += s2[threadIdx.x + o]; }
    __syncthreads();
  }
  float m = s1[0] / (float)C;
  float var = s2[0] / (float)C - m * m;
  float r = rsqrtf(var + 1e-5f);
  for (int k = threadIdx.x; k < C; k += 128)
    y[(long)row * C + k] = (xr[k] - m) * r * g[k] + b[k];
}

__global__ __launch_bounds__(256) void softmax_rows(float* __restrict__ sc,
                                                    const float* __restrict__ mask,
                                                    int Ncol)
{
  __shared__ float red[256];
  float* p = sc + (long)blockIdx.x * Ncol;
  float mx = -3.4e38f;
  for (int j = threadIdx.x; j < Ncol; j += 256) {
    float v = (mask[j] != 0.f) ? p[j] : -3.4e38f;
    mx = fmaxf(mx, v);
  }
  red[threadIdx.x] = mx;
  __syncthreads();
  for (int o = 128; o; o >>= 1) {
    if (threadIdx.x < o) red[threadIdx.x] = fmaxf(red[threadIdx.x], red[threadIdx.x + o]);
    __syncthreads();
  }
  mx = red[0];
  __syncthreads();
  float sum = 0.f;
  for (int j = threadIdx.x; j < Ncol; j += 256) {
    float v = (mask[j] != 0.f) ? __expf(p[j] - mx) : 0.f;
    p[j] = v; sum += v;
  }
  red[threadIdx.x] = sum;
  __syncthreads();
  for (int o = 128; o; o >>= 1) {
    if (threadIdx.x < o) red[threadIdx.x] += red[threadIdx.x + o];
    __syncthreads();
  }
  float inv = 1.f / red[0];
  for (int j = threadIdx.x; j < Ncol; j += 256) p[j] *= inv;
}

__global__ void ew_add(const float* a, const float* b, float* o, int n) {
  int i = blockIdx.x * 256 + threadIdx.x;
  if (i < n) o[i] = a[i] + b[i];
}
__global__ void ew_mul(const float* a, const float* b, float* o, int n) {
  int i = blockIdx.x * 256 + threadIdx.x;
  if (i < n) o[i] = a[i] * b[i];
}

__global__ __launch_bounds__(128) void angle_norm(const float* __restrict__ raw,
                                                  float* __restrict__ ang, int n)
{
  int i = blockIdx.x * 128 + threadIdx.x;
  if (i >= n) return;
  for (int a = 0; a < 7; ++a) {
    float s = raw[i * 14 + 2 * a], c = raw[i * 14 + 2 * a + 1];
    float nrm = sqrtf(fmaxf(s * s + c * c, 1e-8f));
    ang[i * 14 + 2 * a]     = s / nrm;
    ang[i * 14 + 2 * a + 1] = c / nrm;
  }
}

// Per-residue rigid frame construction + torsion frames + atom14 placement.
__global__ __launch_bounds__(128) void geometry(
    const float* __restrict__ nca, const float* __restrict__ ang,
    const int* __restrict__ aatype, const float* __restrict__ def_frames,
    const int* __restrict__ group_idx, const float* __restrict__ atom_mask,
    const float* __restrict__ lit_pos, float* __restrict__ out)
{
  int i = blockIdx.x * 128 + threadIdx.x;
  if (i >= NRES) return;
  const float* q = nca + i * 9;
  float n0 = q[0], n1 = q[1], n2 = q[2];
  float ca0 = q[3], ca1 = q[4], ca2 = q[5];
  float c0 = q[6], c1 = q[7], c2 = q[8];
  // _rigid_from_3pts
  float v10 = c0 - ca0, v11 = c1 - ca1, v12 = c2 - ca2;
  float v20 = n0 - ca0, v21 = n1 - ca1, v22 = n2 - ca2;
  float i1 = rsqrtf(v10 * v10 + v11 * v11 + v12 * v12 + 1e-7f);
  float e10 = v10 * i1, e11 = v11 * i1, e12 = v12 * i1;
  float d = e10 * v20 + e11 * v21 + e12 * v22;
  float u20 = v20 - e10 * d, u21 = v21 - e11 * d, u22 = v22 - e12 * d;
  float i2 = rsqrtf(u20 * u20 + u21 * u21 + u22 * u22 + 1e-7f);
  float e20 = u20 * i2, e21 = u21 * i2, e22 = u22 * i2;
  float e30 = e11 * e22 - e12 * e21;
  float e31 = e12 * e20 - e10 * e22;
  float e32 = e10 * e21 - e11 * e20;
  float R[9] = { e10, e20, e30,  e11, e21, e31,  e12, e22, e32 }; // columns e1,e2,e3
  float t0 = ca0 * 10.f, t1 = ca1 * 10.f, t2 = ca2 * 10.f;        // TRANS_SCALE

  int aa = aatype[i];
  float allR[8][9], allt[8][3];
  for (int g = 0; g < 8; ++g) {
    const float* df = def_frames + ((aa * 8 + g) * 16);
    float dR[9] = { df[0], df[1], df[2], df[4], df[5], df[6], df[8], df[9], df[10] };
    float dt[3] = { df[3], df[7], df[11] };
    float sa, ce;
    if (g == 0) { sa = 0.f; ce = 1.f; }
    else { sa = ang[i * 14 + 2 * (g - 1)]; ce = ang[i * 14 + 2 * (g - 1) + 1]; }
    for (int rr = 0; rr < 3; ++rr) {
      allR[g][rr * 3 + 0] = dR[rr * 3 + 0];
      allR[g][rr * 3 + 1] =  ce * dR[rr * 3 + 1] + sa * dR[rr * 3 + 2];
      allR[g][rr * 3 + 2] = -sa * dR[rr * 3 + 1] + ce * dR[rr * 3 + 2];
    }
    allt[g][0] = dt[0]; allt[g][1] = dt[1]; allt[g][2] = dt[2];
  }
  // chain-compose frames 5,6,7 on top of frame 4 (then 5, then 6)
  for (int g = 5; g < 8; ++g) {
    float Rp[9], tp[3];
    for (int rr = 0; rr < 3; ++rr)
      for (int cc = 0; cc < 3; ++cc)
        Rp[rr * 3 + cc] = allR[g - 1][rr * 3 + 0] * allR[g][0 * 3 + cc]
                        + allR[g - 1][rr * 3 + 1] * allR[g][1 * 3 + cc]
                        + allR[g - 1][rr * 3 + 2] * allR[g][2 * 3 + cc];
    for (int rr = 0; rr < 3; ++rr)
      tp[rr] = allR[g - 1][rr * 3 + 0] * allt[g][0]
             + allR[g - 1][rr * 3 + 1] * allt[g][1]
             + allR[g - 1][rr * 3 + 2] * allt[g][2] + allt[g - 1][rr];
    for (int k2 = 0; k2 < 9; ++k2) allR[g][k2] = Rp[k2];
    for (int k2 = 0; k2 < 3; ++k2) allt[g][k2] = tp[k2];
  }
  // compose with backbone frame
  float gR[8][9], gt[8][3];
  for (int g = 0; g < 8; ++g) {
    for (int rr = 0; rr < 3; ++rr)
      for (int cc = 0; cc < 3; ++cc)
        gR[g][rr * 3 + cc] = R[rr * 3 + 0] * allR[g][0 * 3 + cc]
                           + R[rr * 3 + 1] * allR[g][1 * 3 + cc]
                           + R[rr * 3 + 2] * allR[g][2 * 3 + cc];
    float tt[3] = { t0, t1, t2 };
    for (int rr = 0; rr < 3; ++rr)
      gt[g][rr] = R[rr * 3 + 0] * allt[g][0] + R[rr * 3 + 1] * allt[g][1]
                + R[rr * 3 + 2] * allt[g][2] + tt[rr];
  }
  // atom14 positions
  for (int a = 0; a < 14; ++a) {
    int g = group_idx[aa * 14 + a];
    const float* lp = lit_pos + (aa * 14 + a) * 3;
    float mk = atom_mask[aa * 14 + a];
    for (int rr = 0; rr < 3; ++rr) {
      float p = gR[g][rr * 3 + 0] * lp[0] + gR[g][rr * 3 + 1] * lp[1]
              + gR[g][rr * 3 + 2] * lp[2] + gt[g][rr];
      out[(i * 14 + a) * 3 + rr] = p * mk;
    }
  }
}

// ============================================================================
extern "C" void kernel_launch(void* const* d_in, const int* in_sizes, int n_in,
                              void* d_out, int out_size, void* d_ws, size_t ws_size,
                              hipStream_t stream)
{
  (void)in_sizes; (void)n_in; (void)out_size; (void)ws_size;
  const float* s_in = (const float*)d_in[0];
  const float* z    = (const float*)d_in[1];
  const int*   aaty = (const int*)d_in[2];
  const float* mask = (const float*)d_in[3];
  const float* ln_s_g = (const float*)d_in[4];
  const float* ln_s_b = (const float*)d_in[5];
  const float* ln_z_g = (const float*)d_in[6];
  const float* ln_z_b = (const float*)d_in[7];
  const float* lin_in_w = (const float*)d_in[8];
  const float* lin_in_b = (const float*)d_in[9];
  const float* lin_b_w = (const float*)d_in[10];
  const float* lin_b_b = (const float*)d_in[11];
  const float* bb_w = (const float*)d_in[12];
  const float* bb_b = (const float*)d_in[13];
  const float* ar_in_w = (const float*)d_in[14];
  const float* ar_in_b = (const float*)d_in[15];
  const float* ar_init_w = (const float*)d_in[16];
  const float* ar_init_b = (const float*)d_in[17];
  const float* ar_out_w = (const float*)d_in[18];
  const float* ar_out_b = (const float*)d_in[19];
  // ar_blocks: 20..27  (blk: w1,b1,w2,b2)
  // blocks:    28 + 13*i + {0:ln_g 1:ln_b 2:proj_w 3:o_w 4:o_b 5:g_w 6:g_b
  //                         7:t_ln_g 8:t_ln_b 9:t_w1 10:t_b1 11:t_w2 12:t_b2}
  const float* def_frames = (const float*)d_in[132];
  const int*   group_idx  = (const int*)d_in[133];
  const float* atom_mask  = (const float*)d_in[134];
  const float* lit_pos    = (const float*)d_in[135];

  // workspace allocator (256B aligned)
  char* wbase = (char*)d_ws;
  size_t off = 0;
  auto alloc = [&](size_t nfloats) -> float* {
    float* p = (float*)(wbase + off);
    off = (off + nfloats * 4 + 255) & ~(size_t)255;
    return p;
  };
  float* scores = alloc((size_t)NHEADS * MZ);      // 12 x 512 x 512
  float* sbuf   = alloc(NRES * CS);
  float* sinit  = alloc(NRES * CS);
  float* xn     = alloc(NRES * CS);
  float* qkv    = alloc(NRES * 3 * CS);
  float* ybuf   = alloc(NRES * CS);
  float* gate   = alloc(NRES * CS);
  float* gy     = alloc(NRES * CS);
  float* attn   = alloc(NRES * CS);
  float* h1     = alloc(NRES * CS);
  float* h2     = alloc(NRES * CS);
  float* si     = alloc(NRES * CRES);
  float* abuf   = alloc(NRES * CRES);
  float* tt1    = alloc(NRES * CRES);
  float* tt2    = alloc(NRES * CRES);
  float* araw   = alloc(NRES * 14);
  float* angn   = alloc(NRES * 14);
  float* nca    = alloc(NRES * 9);

  const int NS = NRES * CS;
  const dim3 B256(256), B128(128);
  const float iscale = 0.17677669529663689f;   // 1/sqrt(32)

  // s_initial = LN(s); s = s_initial @ lin_in_w + b
  ln_rows<<<NRES, B128, 0, stream>>>(s_in, ln_s_g, ln_s_b, sinit, CS);
  gemm_wmma<0,0,false,false><<<dim3(32,3,1), B256, 0, stream>>>(
      sinit,0, lin_in_w,0, lin_in_b, nullptr,0,0, nullptr,nullptr,
      sbuf,0, NRES,CS,CS, CS,CS,CS, 1.f);

  for (int i = 0; i < NBLK; ++i) {
    const float* ln_g   = (const float*)d_in[28 + 13*i + 0];
    const float* ln_b   = (const float*)d_in[28 + 13*i + 1];
    const float* proj_w = (const float*)d_in[28 + 13*i + 2];
    const float* o_w    = (const float*)d_in[28 + 13*i + 3];
    const float* o_b    = (const float*)d_in[28 + 13*i + 4];
    const float* g_w    = (const float*)d_in[28 + 13*i + 5];
    const float* g_b    = (const float*)d_in[28 + 13*i + 6];
    const float* t_ln_g = (const float*)d_in[28 + 13*i + 7];
    const float* t_ln_b = (const float*)d_in[28 + 13*i + 8];
    const float* t_w1   = (const float*)d_in[28 + 13*i + 9];
    const float* t_b1   = (const float*)d_in[28 + 13*i + 10];
    const float* t_w2   = (const float*)d_in[28 + 13*i + 11];
    const float* t_b2   = (const float*)d_in[28 + 13*i + 12];

    // scores[h][i*512+j] = LN(z[i,j]) @ lin_b_w[:, i*12+h] + lin_b_b  (transposed store)
    gemm_wmma<2,0,false,true><<<dim3(MZ/16,1,1), B256, 0, stream>>>(
        z,0, lin_b_w + i*NHEADS, 0, lin_b_b + i*NHEADS,
        nullptr,0,0, ln_z_g, ln_z_b,
        scores,0, MZ,CZ,NHEADS, CZ, NBLK*NHEADS, MZ, 1.f);

    // xn = LN(s); qkv = xn @ proj_w
    ln_rows<<<NRES, B128, 0, stream>>>(sbuf, ln_g, ln_b, xn, CS);
    gemm_wmma<0,0,false,false><<<dim3(32,9,1), B256, 0, stream>>>(
        xn,0, proj_w,0, nullptr, nullptr,0,0, nullptr,nullptr,
        qkv,0, NRES,CS,3*CS, CS,3*CS,3*CS, 1.f);

    // scores = q@k^T / sqrt(32) + bias   (B transposed; batched over heads)
    gemm_wmma<0,0,true,false><<<dim3(32,4,NHEADS), B256, 0, stream>>>(
        qkv, 3*HDIM, qkv + HDIM, 3*HDIM, nullptr,
        scores, (long)MZ, NRES, nullptr,nullptr,
        scores, (long)MZ, NRES,HDIM,NRES, 3*CS,3*CS,NRES, iscale);

    softmax_rows<<<NHEADS*NRES, B256, 0, stream>>>(scores, mask, NRES);

    // y[:, h*32 + c] = P_h @ V_h   (batched over heads)
    gemm_wmma<0,0,false,false><<<dim3(32,1,NHEADS), B256, 0, stream>>>(
        scores, (long)MZ, qkv + 2*HDIM, 3*HDIM, nullptr,
        nullptr,0,0, nullptr,nullptr,
        ybuf, (long)HDIM, NRES,NRES,HDIM, NRES,3*CS,CS, 1.f);

    // gate = sigmoid(xn @ g_w + g_b); attn = (gate*y) @ o_w + o_b; s += attn
    gemm_wmma<0,2,false,false><<<dim3(32,3,1), B256, 0, stream>>>(
        xn,0, g_w,0, g_b, nullptr,0,0, nullptr,nullptr,
        gate,0, NRES,CS,CS, CS,CS,CS, 1.f);
    ew_mul<<<(NS+255)/256, B256, 0, stream>>>(gate, ybuf, gy, NS);
    gemm_wmma<0,0,false,false><<<dim3(32,3,1), B256, 0, stream>>>(
        gy,0, o_w,0, o_b, nullptr,0,0, nullptr,nullptr,
        attn,0, NRES,CS,CS, CS,CS,CS, 1.f);
    ew_add<<<(NS+255)/256, B256, 0, stream>>>(sbuf, attn, sbuf, NS);

    // transition: s += relu(LN(s)@t_w1+t_b1) @ t_w2 + t_b2
    gemm_wmma<2,1,false,false><<<dim3(32,3,1), B256, 0, stream>>>(
        sbuf,0, t_w1,0, t_b1, nullptr,0,0, t_ln_g, t_ln_b,
        h1,0, NRES,CS,CS, CS,CS,CS, 1.f);
    gemm_wmma<0,0,false,false><<<dim3(32,3,1), B256, 0, stream>>>(
        h1,0, t_w2,0, t_b2, nullptr,0,0, nullptr,nullptr,
        h2,0, NRES,CS,CS, CS,CS,CS, 1.f);
    ew_add<<<(NS+255)/256, B256, 0, stream>>>(sbuf, h2, sbuf, NS);
  }

  // angle resnet
  gemm_wmma<1,0,false,false><<<dim3(32,1,1), B256, 0, stream>>>(
      sinit,0, ar_init_w,0, ar_init_b, nullptr,0,0, nullptr,nullptr,
      si,0, NRES,CS,CRES, CS,CRES,CRES, 1.f);
  gemm_wmma<1,0,false,false><<<dim3(32,1,1), B256, 0, stream>>>(
      sbuf,0, ar_in_w,0, ar_in_b, si,0,CRES, nullptr,nullptr,
      abuf,0, NRES,CS,CRES, CS,CRES,CRES, 1.f);
  for (int b = 0; b < 2; ++b) {
    const float* w1 = (const float*)d_in[20 + 4*b + 0];
    const float* b1 = (const float*)d_in[20 + 4*b + 1];
    const float* w2 = (const float*)d_in[20 + 4*b + 2];
    const float* b2 = (const float*)d_in[20 + 4*b + 3];
    gemm_wmma<1,0,false,false><<<dim3(32,1,1), B256, 0, stream>>>(
        abuf,0, w1,0, b1, nullptr,0,0, nullptr,nullptr,
        tt1,0, NRES,CRES,CRES, CRES,CRES,CRES, 1.f);
    gemm_wmma<1,0,false,false><<<dim3(32,1,1), B256, 0, stream>>>(
        tt1,0, w2,0, b2, nullptr,0,0, nullptr,nullptr,
        tt2,0, NRES,CRES,CRES, CRES,CRES,CRES, 1.f);
    ew_add<<<(NRES*CRES+255)/256, B256, 0, stream>>>(abuf, tt2, abuf, NRES*CRES);
  }
  gemm_wmma<1,0,false,false><<<dim3(32,1,1), B256, 0, stream>>>(
      abuf,0, ar_out_w,0, ar_out_b, nullptr,0,0, nullptr,nullptr,
      araw,0, NRES,CRES,14, CRES,14,14, 1.f);
  angle_norm<<<4, B128, 0, stream>>>(araw, angn, NRES);

  // backbone frame inputs + geometry
  gemm_wmma<0,0,false,false><<<dim3(32,1,1), B256, 0, stream>>>(
      sbuf,0, bb_w,0, bb_b, nullptr,0,0, nullptr,nullptr,
      nca,0, NRES,CS,9, CS,9,9, 1.f);
  geometry<<<4, B128, 0, stream>>>(nca, angn, aaty, def_frames, group_idx,
                                   atom_mask, lit_pos, (float*)d_out);
}
